// DockingHead_75118978007444
// MI455X (gfx1250) — matrix operand order repaired and checked by reading, not compiled, not run on hardware
//
#include <hip/hip_runtime.h>

typedef __bf16 bf16;
typedef bf16  v16bf __attribute__((ext_vector_type(16)));
typedef float v8f   __attribute__((ext_vector_type(8)));

#define N_ATOM 8192
#define N_PROT 16384
#define N_FRAG 1024
#define KNN    32
#define E_RA   (N_ATOM * KNN)
#define SDIM   128
#define VDIM   64
#define TDIM   128
#define ESP    416      // 400 padded to 13*32
#define HIDD   256
#define OUTC   649      // 128 + 192 + 128 + 192 + 9

// ---------------- WMMA fragment helpers ----------------
union FragU { v16bf v; uint4 q[2]; };

// A fragment (16x32 bf16, MxK) from an LDS tile, row-major, stride in bf16 elems.
__device__ __forceinline__ v16bf load_a_tile(const bf16* tile, int stride,
                                             int rowBase, int kBase, int lane) {
  int row = rowBase + (lane & 15);
  int k0  = kBase + ((lane >> 4) << 3);
  const bf16* p = tile + row * stride + k0;
  FragU f;
  f.q[0] = *(const uint4*)(p);
  f.q[1] = *(const uint4*)(p + 16);
  return f.v;
}

// B fragment (32x16 bf16, KxN) from pre-swizzled blocks (512 bf16 per block,
// lane-major, 16 contiguous bf16 per lane). Works for global or LDS pointers.
__device__ __forceinline__ v16bf load_b_sw(const bf16* Wsw, int ntilesN,
                                           int kk, int nt, int lane) {
  const bf16* p = Wsw + (((size_t)(kk * ntilesN + nt)) << 9) + (lane << 4);
  FragU f;
  f.q[0] = *(const uint4*)(p);
  f.q[1] = *(const uint4*)(p + 8);
  return f.v;
}

__device__ __forceinline__ v8f wmma_bf16(v16bf a, v16bf b, v8f c) {
  return __builtin_amdgcn_wmma_f32_16x16x32_bf16(false, a, false, b, (short)0, c,
                                                 false, false);
}

__device__ __forceinline__ float silu_f(float x) {
  return x * __builtin_amdgcn_rcpf(1.f + __expf(-x));
}

// ---------------- CDNA5 async LDS staging ----------------
// Low 32 bits of a generic pointer into the LDS aperture are the DS-space offset.
__device__ __forceinline__ unsigned lds_off_u32(const void* p) {
  return (unsigned)(size_t)p;
}
// GLOBAL_LOAD_ASYNC_TO_LDS_B128: per-lane 16B memory->LDS copy, tracked by ASYNCcnt.
__device__ __forceinline__ void async_copy_b128(unsigned lds_off, const void* gptr) {
  asm volatile("global_load_async_to_lds_b128 %0, %1, off"
               :: "v"(lds_off), "v"(gptr) : "memory");
}
__device__ __forceinline__ void wait_async0() {
  asm volatile("s_wait_asynccnt 0x0" ::: "memory");
}

// ---------------- weight swizzle prep ----------------
__global__ __launch_bounds__(256) void swizzle_kernel(const float* __restrict__ W,
                                                      int K, int N,
                                                      bf16* __restrict__ out, int total) {
  int i = blockIdx.x * blockDim.x + threadIdx.x;
  if (i >= total) return;
  int b = i >> 9, within = i & 511;
  int lane = within >> 4, elem = within & 15;
  int ntn = N >> 4;
  int kk = b / ntn, nt = b - kk * ntn;
  int k = kk * 32 + ((lane >> 4) << 3) + elem + ((elem >> 3) << 3);
  int n = nt * 16 + (lane & 15);
  float v = (k < K) ? W[(size_t)k * N + n] : 0.f;
  out[i] = (bf16)v;
}

// ---------------- KNN (batch-restricted radius topk) ----------------
__global__ __launch_bounds__(256) void knn_kernel(const float* __restrict__ atom_pos,
                                                  const float* __restrict__ prot_pos,
                                                  const int* __restrict__ atom_batch,
                                                  const int* __restrict__ prot_batch,
                                                  int* __restrict__ srcIdx,
                                                  float* __restrict__ validF) {
  int i = blockIdx.x * blockDim.x + threadIdx.x;
  if (i >= N_ATOM) return;
  float ax = atom_pos[i * 3 + 0], ay = atom_pos[i * 3 + 1], az = atom_pos[i * 3 + 2];
  int b = atom_batch[i];
  int lo = 0, hi = N_PROT;
  while (lo < hi) { int m = (lo + hi) >> 1; if (prot_batch[m] < b) lo = m + 1; else hi = m; }
  int s0 = lo; hi = N_PROT;
  while (lo < hi) { int m = (lo + hi) >> 1; if (prot_batch[m] <= b) lo = m + 1; else hi = m; }
  int s1 = lo;
  int fallback = (s0 < N_PROT - 1) ? s0 : (N_PROT - 1);
  float bd[KNN]; int bi[KNN];
#pragma unroll
  for (int k = 0; k < KNN; ++k) { bd[k] = 3.0e38f; bi[k] = fallback; }
  float wmax = 3.0e38f; int wslot = 0;
  for (int j = s0; j < s1; ++j) {
    float dx = ax - prot_pos[j * 3 + 0];
    float dy = ay - prot_pos[j * 3 + 1];
    float dz = az - prot_pos[j * 3 + 2];
    float d = sqrtf(dx * dx + dy * dy + dz * dz);
    if (d <= 3.0f && d < wmax) {
      bd[wslot] = d; bi[wslot] = j;
      wmax = -1.f;
      for (int k = 0; k < KNN; ++k) if (bd[k] > wmax) { wmax = bd[k]; wslot = k; }
    }
  }
  for (int k = 0; k < KNN; ++k) {
    srcIdx[(size_t)i * KNN + k] = bi[k];
    validF[(size_t)i * KNN + k] = (bd[k] < 1.0e38f) ? 1.f : 0.f;
  }
}

// ---------------- fused TP-conv (64 edges per block, 4 waves) ----------------
#define L_ES    0         // 64*416*2 = 53248
#define L_HT    53248     // 64*256*2 = 32768
#define L_GT    86016     // 64*256*2 = 32768
#define L_HV    118784    // 64*192*2 = 24576
#define L_VALID 143360
#define L_DIST  143616
#define L_UNIT  143872    // 64*3 f32
#define L_SRC   144640
#define L_FEAT  144896
#define L_AIDX  145152
#define L_DST   145408
#define L_PS    145664    // 4*128 f32
#define L_PV    147712    // 4*192 f32
#define L_WV    150784    // 8192 B: async-staged Wv fragments
#define CONV_LDS 158976   // 2 WGs still fit a 320KB WGP

template <bool RA>
__global__ __launch_bounds__(128) void conv_kernel(
    const float* __restrict__ atom_pos,
    const float* __restrict__ other_pos,   // RA: prot_pos, AF: T_frag
    const int* __restrict__ srcIdx,        // RA only
    const float* __restrict__ validF,      // RA only
    const int* __restrict__ frag_id,       // AF only
    const float* __restrict__ hs_src,      // RA: h_prot_s, AF: ha_s
    const float* __restrict__ hv_src,      // RA: h_prot_v, AF: ha_v
    const float* __restrict__ featB,       // RA: h_atom_s, AF: h_frag_s
    const float* __restrict__ temb,
    const bf16* __restrict__ W1sw, const float* __restrict__ b1,
    const bf16* __restrict__ W2sw, const float* __restrict__ b2,
    const bf16* __restrict__ Wssw, const bf16* __restrict__ Wvsw,
    float* __restrict__ agg_s, float* __restrict__ agg_v) {
  extern __shared__ __align__(32) char smem[];
  bf16*  esT    = (bf16*)(smem + L_ES);
  bf16*  hT     = (bf16*)(smem + L_HT);
  bf16*  gT     = (bf16*)(smem + L_GT);
  bf16*  hvT    = (bf16*)(smem + L_HV);
  float* validT = (float*)(smem + L_VALID);
  float* distT  = (float*)(smem + L_DIST);
  float* unitT  = (float*)(smem + L_UNIT);
  int*   srcT   = (int*)(smem + L_SRC);
  int*   featT  = (int*)(smem + L_FEAT);
  int*   aT     = (int*)(smem + L_AIDX);
  int*   dstT   = (int*)(smem + L_DST);
  float* pS     = (float*)(smem + L_PS);
  float* pV     = (float*)(smem + L_PV);
  bf16*  wvT    = (bf16*)(smem + L_WV);

  int tid = threadIdx.x;
  int e0 = blockIdx.x * 64;

  // ---- async-stage the shared Wv fragment blocks (8KB) into LDS ----
  {
    unsigned dst0 = lds_off_u32(wvT);
    for (int t = tid; t < 512; t += 128)   // 512 x 16B
      async_copy_b128(dst0 + t * 16, (const char*)Wvsw + t * 16);
  }

  // ---- phase 1a: per-row metadata ----
  if (tid < 64) {
    int e = e0 + tid;
    int aIdx, sIdx, fIdx; float vld;
    if (RA) { aIdx = e >> 5; sIdx = srcIdx[e]; fIdx = aIdx; vld = validF[e]; }
    else    { aIdx = e;      sIdx = e;         fIdx = frag_id[e]; vld = 1.f; }
    int oIdx = RA ? sIdx : fIdx;
    float dx = atom_pos[aIdx * 3 + 0] - other_pos[oIdx * 3 + 0];
    float dy = atom_pos[aIdx * 3 + 1] - other_pos[oIdx * 3 + 1];
    float dz = atom_pos[aIdx * 3 + 2] - other_pos[oIdx * 3 + 2];
    float d = sqrtf(dx * dx + dy * dy + dz * dz);
    float inv = __builtin_amdgcn_rcpf(d + 1e-8f);
    validT[tid] = vld; distT[tid] = d;
    unitT[tid * 3 + 0] = dx * inv;
    unitT[tid * 3 + 1] = dy * inv;
    unitT[tid * 3 + 2] = dz * inv;
    srcT[tid] = sIdx; featT[tid] = fIdx; aT[tid] = aIdx;
    dstT[tid] = RA ? aIdx : fIdx;
  }
  __syncthreads();

  // ---- phase 1b: build es tile [64 x 416] and hv tile [64 x (3*64)] in bf16 ----
  for (int idx = tid; idx < 64 * ESP; idx += 128) {
    int r = idx / ESP, c = idx - r * ESP;
    float v;
    if (c < 16) {                                 // RBF
      float t = 2.66666667f * (distT[r] - 0.4f * (float)c);
      v = __expf(-t * t);
    } else if (c < 144) v = hs_src[(size_t)srcT[r]  * SDIM + (c - 16)];
    else if (c < 272)   v = featB [(size_t)featT[r] * SDIM + (c - 144)];
    else if (c < 400)   v = temb  [(size_t)aT[r]    * TDIM + (c - 272)];
    else v = 0.f;
    esT[idx] = (bf16)v;
  }
  for (int idx = tid; idx < 64 * 192; idx += 128) {
    int r = idx / 192, j = idx - r * 192;
    int c3 = j >> 6, v = j & 63;
    hvT[idx] = (bf16)hv_src[(size_t)srcT[r] * 192 + v * 3 + c3];
  }
  wait_async0();
  __syncthreads();

  int w = tid >> 5, lane = tid & 31;
  int rowBase = w << 4;
  int colLane = lane & 15;
  int rOff = (lane >> 4) << 3;
  const v8f z8 = {0.f, 0.f, 0.f, 0.f, 0.f, 0.f, 0.f, 0.f};

  // ---- GEMM1: h = silu(es @ W1 + b1), bf16 into hT (same-wave rows only) ----
  for (int nt = 0; nt < 16; ++nt) {
    v8f acc = z8;
#pragma unroll
    for (int kk = 0; kk < 13; ++kk) {
      if (kk + 1 < 13)
        __builtin_prefetch(W1sw + (((size_t)((kk + 1) * 16 + nt)) << 9), 0, 0);
      acc = wmma_bf16(load_a_tile(esT, ESP, rowBase, kk * 32, lane),
                      load_b_sw(W1sw, 16, kk, nt, lane), acc);
    }
    int col = (nt << 4) + colLane;
    float bb = b1[col];
#pragma unroll
    for (int r = 0; r < 8; ++r) {
      int row = rowBase + rOff + r;
      hT[row * HIDD + col] = (bf16)silu_f(acc[r] + bb);
    }
  }

  // ---- GEMM2: gate = h @ W2 + b2, bf16 into gT ----
  for (int nt = 0; nt < 16; ++nt) {
    v8f acc = z8;
#pragma unroll
    for (int kk = 0; kk < 8; ++kk) {
      if (kk + 1 < 8)
        __builtin_prefetch(W2sw + (((size_t)((kk + 1) * 16 + nt)) << 9), 0, 0);
      acc = wmma_bf16(load_a_tile(hT, HIDD, rowBase, kk * 32, lane),
                      load_b_sw(W2sw, 16, kk, nt, lane), acc);
    }
    int col = (nt << 4) + colLane;
    float bb = b2[col];
#pragma unroll
    for (int r = 0; r < 8; ++r) {
      int row = rowBase + rOff + r;
      gT[row * HIDD + col] = (bf16)(acc[r] + bb);
    }
  }

  // ---- xs = hs_src @ Ws (es cols 16..143), gated scalar message + scatter ----
  for (int nt = 0; nt < 8; ++nt) {
    v8f acc = z8;
#pragma unroll
    for (int kk = 0; kk < 4; ++kk)
      acc = wmma_bf16(load_a_tile(esT, ESP, rowBase, 16 + kk * 32, lane),
                      load_b_sw(Wssw, 8, kk, nt, lane), acc);
    int col = (nt << 4) + colLane;
    float sum = 0.f;
#pragma unroll
    for (int r = 0; r < 8; ++r) {
      int row = rowBase + rOff + r;
      float val = acc[r] * (float)gT[row * HIDD + col] * validT[row];
      if (RA) sum += val;
      else atomicAdd(&agg_s[(size_t)dstT[row] * SDIM + col], val);
    }
    if (RA) {
      sum += __shfl_xor(sum, 16, 32);
      if (lane < 16) pS[w * SDIM + col] = sum;
    }
  }

  // ---- xv = hv_src @ Wv per coord (B from async-staged LDS) ----
  for (int c3 = 0; c3 < 3; ++c3) {
    for (int nt = 0; nt < 4; ++nt) {
      v8f acc = z8;
#pragma unroll
      for (int kk = 0; kk < 2; ++kk)
        acc = wmma_bf16(load_a_tile(hvT, 192, rowBase, c3 * 64 + kk * 32, lane),
                        load_b_sw(wvT, 4, kk, nt, lane), acc);
      int vcol = (nt << 4) + colLane;
      float sum = 0.f;
#pragma unroll
      for (int r = 0; r < 8; ++r) {
        int row = rowBase + rOff + r;
        float g1 = (float)gT[row * HIDD + SDIM + vcol];
        float g2 = (float)gT[row * HIDD + SDIM + VDIM + vcol];
        float val = (acc[r] * g1 + g2 * unitT[row * 3 + c3]) * validT[row];
        if (RA) sum += val;
        else atomicAdd(&agg_v[(size_t)dstT[row] * 192 + vcol * 3 + c3], val);
      }
      if (RA) {
        sum += __shfl_xor(sum, 16, 32);
        if (lane < 16) pV[w * 192 + vcol * 3 + c3] = sum;
      }
    }
  }

  // ---- RA: combine per-wave partials -> 2 atoms per block, direct store ----
  if (RA) {
    __syncthreads();
    int atom0 = e0 >> 5;
    for (int idx = tid; idx < 2 * SDIM; idx += 128) {
      int a = idx >> 7, col = idx & 127;
      agg_s[(size_t)(atom0 + a) * SDIM + col] =
          pS[(2 * a) * SDIM + col] + pS[(2 * a + 1) * SDIM + col];
    }
    for (int idx = tid; idx < 2 * 192; idx += 128) {
      int a = idx / 192, j = idx - a * 192;
      agg_v[(size_t)(atom0 + a) * 192 + j] =
          pV[(2 * a) * 192 + j] + pV[(2 * a + 1) * 192 + j];
    }
  }
}

// ---------------- post: out_s = base_s + silu(agg_s@Ws); out_v = base_v + agg_v@Wv ----
#define POST_LDS 81920   // 16K sT + 24K vT + 32K Ws + 8K Wv
__global__ __launch_bounds__(128) void post_kernel(
    int M, const float* __restrict__ base_s, const float* __restrict__ base_v,
    const float* __restrict__ agg_s, const float* __restrict__ agg_v,
    const bf16* __restrict__ Wssw, const bf16* __restrict__ Wvsw,
    float* __restrict__ out_s, float* __restrict__ out_v) {
  extern __shared__ __align__(32) char smem[];
  bf16* sT  = (bf16*)(smem);
  bf16* vT  = (bf16*)(smem + 16384);
  bf16* wsT = (bf16*)(smem + 40960);
  bf16* wvT = (bf16*)(smem + 73728);
  int tid = threadIdx.x;
  int r0 = blockIdx.x * 64;
  if (r0 >= M) return;
  // async-stage both weight matrices while A tiles are gathered/converted
  {
    unsigned ds0 = lds_off_u32(wsT);
    for (int t = tid; t < 2048; t += 128)
      async_copy_b128(ds0 + t * 16, (const char*)Wssw + t * 16);
    unsigned ds1 = lds_off_u32(wvT);
    for (int t = tid; t < 512; t += 128)
      async_copy_b128(ds1 + t * 16, (const char*)Wvsw + t * 16);
  }
  for (int idx = tid; idx < 64 * 128; idx += 128) {
    int r = idx >> 7, c = idx & 127;
    sT[idx] = (bf16)agg_s[(size_t)(r0 + r) * 128 + c];
  }
  for (int idx = tid; idx < 64 * 192; idx += 128) {
    int r = idx / 192, j = idx - r * 192;
    int c3 = j >> 6, v = j & 63;
    vT[idx] = (bf16)agg_v[(size_t)(r0 + r) * 192 + v * 3 + c3];
  }
  wait_async0();
  __syncthreads();
  int w = tid >> 5, lane = tid & 31;
  int rowBase = w << 4, colLane = lane & 15, rOff = (lane >> 4) << 3;
  const v8f z8 = {0.f, 0.f, 0.f, 0.f, 0.f, 0.f, 0.f, 0.f};
  for (int nt = 0; nt < 8; ++nt) {
    v8f acc = z8;
#pragma unroll
    for (int kk = 0; kk < 4; ++kk)
      acc = wmma_bf16(load_a_tile(sT, 128, rowBase, kk * 32, lane),
                      load_b_sw(wsT, 8, kk, nt, lane), acc);
    int col = (nt << 4) + colLane;
#pragma unroll
    for (int r = 0; r < 8; ++r) {
      int rowG = r0 + rowBase + rOff + r;
      out_s[(size_t)rowG * 128 + col] =
          base_s[(size_t)rowG * 128 + col] + silu_f(acc[r]);
    }
  }
  for (int c3 = 0; c3 < 3; ++c3) {
    for (int nt = 0; nt < 4; ++nt) {
      v8f acc = z8;
#pragma unroll
      for (int kk = 0; kk < 2; ++kk)
        acc = wmma_bf16(load_a_tile(vT, 192, rowBase, c3 * 64 + kk * 32, lane),
                        load_b_sw(wvT, 4, kk, nt, lane), acc);
      int vcol = (nt << 4) + colLane;
#pragma unroll
      for (int r = 0; r < 8; ++r) {
        int rowG = r0 + rowBase + rOff + r;
        size_t o = (size_t)rowG * 192 + vcol * 3 + c3;
        out_v[o] = base_v[o] + acc[r];
      }
    }
  }
}

// ---------------- fragment centers + inertia ----------------
__global__ __launch_bounds__(256) void frag_accum_kernel(
    const float* __restrict__ atom_pos, const float* __restrict__ local_pos,
    const int* __restrict__ frag_id, float* __restrict__ fragSum,
    float* __restrict__ fragCnt, float* __restrict__ Iacc) {
  int i = blockIdx.x * blockDim.x + threadIdx.x;
  if (i >= N_ATOM) return;
  int f = frag_id[i];
  atomicAdd(&fragSum[f * 3 + 0], atom_pos[i * 3 + 0]);
  atomicAdd(&fragSum[f * 3 + 1], atom_pos[i * 3 + 1]);
  atomicAdd(&fragSum[f * 3 + 2], atom_pos[i * 3 + 2]);
  atomicAdd(&fragCnt[f], 1.f);
  float x = local_pos[i * 3 + 0], y = local_pos[i * 3 + 1], z = local_pos[i * 3 + 2];
  float r2 = x * x + y * y + z * z;
  atomicAdd(&Iacc[f * 9 + 0], r2 - x * x);
  atomicAdd(&Iacc[f * 9 + 1], -x * y);
  atomicAdd(&Iacc[f * 9 + 2], -x * z);
  atomicAdd(&Iacc[f * 9 + 3], -y * x);
  atomicAdd(&Iacc[f * 9 + 4], r2 - y * y);
  atomicAdd(&Iacc[f * 9 + 5], -y * z);
  atomicAdd(&Iacc[f * 9 + 6], -z * x);
  atomicAdd(&Iacc[f * 9 + 7], -z * y);
  atomicAdd(&Iacc[f * 9 + 8], r2 - z * z);
}

__global__ __launch_bounds__(256) void frag_final_kernel(
    const float* __restrict__ fragSum, const float* __restrict__ fragCnt,
    const float* __restrict__ Iacc, float* __restrict__ T_frag,
    float* __restrict__ Ifin) {
  int f = blockIdx.x * blockDim.x + threadIdx.x;
  if (f >= N_FRAG) return;
  float rc = __builtin_amdgcn_rcpf(fmaxf(fragCnt[f], 1.f));
  T_frag[f * 3 + 0] = fragSum[f * 3 + 0] * rc;
  T_frag[f * 3 + 1] = fragSum[f * 3 + 1] * rc;
  T_frag[f * 3 + 2] = fragSum[f * 3 + 2] * rc;
  float tr = Iacc[f * 9 + 0] + Iacc[f * 9 + 4] + Iacc[f * 9 + 8];
  float add = 1e-4f * fmaxf(tr, 1.f);
  for (int j = 0; j < 9; ++j) {
    float v = Iacc[f * 9 + j];
    if (j == 0 || j == 4 || j == 8) v += add;
    Ifin[f * 9 + j] = v;
  }
}

// ---------------- final assemble ----------------
__global__ __launch_bounds__(256) void assemble_kernel(
    const float* __restrict__ ha_s, const float* __restrict__ ha_v,
    const float* __restrict__ hf_s, const float* __restrict__ hf_v,
    const float* __restrict__ Ifin, const int* __restrict__ frag_id,
    float* __restrict__ out) {
  int i = blockIdx.x * blockDim.x + threadIdx.x;
  if (i >= N_ATOM * OUTC) return;
  int row = i / OUTC, c = i - row * OUTC;
  float v;
  if (c < 128) v = ha_s[(size_t)row * 128 + c];
  else if (c < 320) v = ha_v[(size_t)row * 192 + (c - 128)];
  else {
    int f = frag_id[row];
    if (c < 448) v = hf_s[(size_t)f * 128 + (c - 320)];
    else if (c < 640) v = hf_v[(size_t)f * 192 + (c - 448)];
    else v = Ifin[(size_t)f * 9 + (c - 640)];
  }
  out[i] = v;
}

// ---------------- host ----------------
extern "C" void kernel_launch(void* const* d_in, const int* in_sizes, int n_in,
                              void* d_out, int out_size, void* d_ws, size_t ws_size,
                              hipStream_t stream) {
  const float* atom_pos  = (const float*)d_in[0];
  const float* prot_pos  = (const float*)d_in[1];
  const float* local_pos = (const float*)d_in[2];
  const float* h_atom_s  = (const float*)d_in[3];
  const float* h_atom_v  = (const float*)d_in[4];
  const float* h_prot_s  = (const float*)d_in[5];
  const float* h_prot_v  = (const float*)d_in[6];
  const float* h_frag_s  = (const float*)d_in[7];
  const float* h_frag_v  = (const float*)d_in[8];
  const float* t_emb     = (const float*)d_in[9];
  const float* W1_ra = (const float*)d_in[10]; const float* b1_ra = (const float*)d_in[11];
  const float* W2_ra = (const float*)d_in[12]; const float* b2_ra = (const float*)d_in[13];
  const float* Ws_ra = (const float*)d_in[14]; const float* Wv_ra = (const float*)d_in[15];
  const float* W1_af = (const float*)d_in[16]; const float* b1_af = (const float*)d_in[17];
  const float* W2_af = (const float*)d_in[18]; const float* b2_af = (const float*)d_in[19];
  const float* Ws_af = (const float*)d_in[20]; const float* Wv_af = (const float*)d_in[21];
  const float* Wo_s  = (const float*)d_in[22]; const float* Wo_v  = (const float*)d_in[23];
  const float* Wf_s  = (const float*)d_in[24]; const float* Wf_v  = (const float*)d_in[25];
  const int* frag_id    = (const int*)d_in[26];
  const int* atom_batch = (const int*)d_in[27];
  const int* prot_batch = (const int*)d_in[28];
  float* out = (float*)d_out;

  // ---- workspace carve ----
  char* base = (char*)d_ws;
  size_t off = 0;
  auto alloc = [&](size_t bytes) -> char* {
    char* p = base + off;
    off += (bytes + 255) & ~(size_t)255;
    return p;
  };
  bf16* W1ra_sw = (bf16*)alloc(106496 * 2);
  bf16* W2ra_sw = (bf16*)alloc(65536 * 2);
  bf16* Wsra_sw = (bf16*)alloc(16384 * 2);
  bf16* Wvra_sw = (bf16*)alloc(4096 * 2);
  bf16* W1af_sw = (bf16*)alloc(106496 * 2);
  bf16* W2af_sw = (bf16*)alloc(65536 * 2);
  bf16* Wsaf_sw = (bf16*)alloc(16384 * 2);
  bf16* Wvaf_sw = (bf16*)alloc(4096 * 2);
  bf16* Wos_sw  = (bf16*)alloc(16384 * 2);
  bf16* Wov_sw  = (bf16*)alloc(4096 * 2);
  bf16* Wfs_sw  = (bf16*)alloc(16384 * 2);
  bf16* Wfv_sw  = (bf16*)alloc(4096 * 2);
  int*   srcIdx = (int*)alloc((size_t)E_RA * 4);
  float* validF = (float*)alloc((size_t)E_RA * 4);
  float* agg_s  = (float*)alloc((size_t)N_ATOM * 128 * 4);
  float* agg_v  = (float*)alloc((size_t)N_ATOM * 192 * 4);
  float* ha_s   = (float*)alloc((size_t)N_ATOM * 128 * 4);
  float* ha_v   = (float*)alloc((size_t)N_ATOM * 192 * 4);
  float* fs     = (float*)alloc((size_t)N_FRAG * 128 * 4);
  float* fv     = (float*)alloc((size_t)N_FRAG * 192 * 4);
  float* hf_s   = (float*)alloc((size_t)N_FRAG * 128 * 4);
  float* hf_v   = (float*)alloc((size_t)N_FRAG * 192 * 4);
  float* fragSum = (float*)alloc((size_t)N_FRAG * 3 * 4);
  float* fragCnt = (float*)alloc((size_t)N_FRAG * 4);
  float* T_frag  = (float*)alloc((size_t)N_FRAG * 3 * 4);
  float* Iacc    = (float*)alloc((size_t)N_FRAG * 9 * 4);
  float* Ifin    = (float*)alloc((size_t)N_FRAG * 9 * 4);

  // allow large dynamic LDS
  hipFuncSetAttribute((const void*)conv_kernel<true>,
                      hipFuncAttributeMaxDynamicSharedMemorySize, CONV_LDS);
  hipFuncSetAttribute((const void*)conv_kernel<false>,
                      hipFuncAttributeMaxDynamicSharedMemorySize, CONV_LDS);
  hipFuncSetAttribute((const void*)post_kernel,
                      hipFuncAttributeMaxDynamicSharedMemorySize, POST_LDS);

  // ---- zero accumulators (AF scatter + fragment stats) ----
  hipMemsetAsync(fs, 0, (size_t)N_FRAG * 128 * 4, stream);
  hipMemsetAsync(fv, 0, (size_t)N_FRAG * 192 * 4, stream);
  hipMemsetAsync(fragSum, 0, (size_t)N_FRAG * 3 * 4, stream);
  hipMemsetAsync(fragCnt, 0, (size_t)N_FRAG * 4, stream);
  hipMemsetAsync(Iacc, 0, (size_t)N_FRAG * 9 * 4, stream);

  // ---- swizzle all weights to bf16 fragment order ----
  auto sw = [&](const float* W, int K, int Kp, int N, bf16* dst) {
    int total = (Kp / 32) * (N / 16) * 512;
    swizzle_kernel<<<(total + 255) / 256, 256, 0, stream>>>(W, K, N, dst, total);
  };
  sw(W1_ra, 400, 416, 256, W1ra_sw);
  sw(W2_ra, 256, 256, 256, W2ra_sw);
  sw(Ws_ra, 128, 128, 128, Wsra_sw);
  sw(Wv_ra, 64, 64, 64, Wvra_sw);
  sw(W1_af, 400, 416, 256, W1af_sw);
  sw(W2_af, 256, 256, 256, W2af_sw);
  sw(Ws_af, 128, 128, 128, Wsaf_sw);
  sw(Wv_af, 64, 64, 64, Wvaf_sw);
  sw(Wo_s, 128, 128, 128, Wos_sw);
  sw(Wo_v, 64, 64, 64, Wov_sw);
  sw(Wf_s, 128, 128, 128, Wfs_sw);
  sw(Wf_v, 64, 64, 64, Wfv_sw);

  // ---- KNN graph ----
  knn_kernel<<<N_ATOM / 256, 256, 0, stream>>>(atom_pos, prot_pos, atom_batch,
                                               prot_batch, srcIdx, validF);

  // ---- fragment centers + inertia (independent of convs) ----
  frag_accum_kernel<<<N_ATOM / 256, 256, 0, stream>>>(atom_pos, local_pos, frag_id,
                                                      fragSum, fragCnt, Iacc);
  frag_final_kernel<<<N_FRAG / 256, 256, 0, stream>>>(fragSum, fragCnt, Iacc,
                                                      T_frag, Ifin);

  // ---- prot -> atom TP conv (fused, LDS-resident) ----
  conv_kernel<true><<<E_RA / 64, 128, CONV_LDS, stream>>>(
      atom_pos, prot_pos, srcIdx, validF, nullptr,
      h_prot_s, h_prot_v, h_atom_s, t_emb,
      W1ra_sw, b1_ra, W2ra_sw, b2_ra, Wsra_sw, Wvra_sw,
      agg_s, agg_v);

  // ---- ha update ----
  post_kernel<<<N_ATOM / 64, 128, POST_LDS, stream>>>(
      N_ATOM, h_atom_s, h_atom_v, agg_s, agg_v, Wos_sw, Wov_sw, ha_s, ha_v);

  // ---- atom -> frag TP conv (atomic scatter) ----
  conv_kernel<false><<<N_ATOM / 64, 128, CONV_LDS, stream>>>(
      atom_pos, T_frag, nullptr, nullptr, frag_id,
      ha_s, ha_v, h_frag_s, t_emb,
      W1af_sw, b1_af, W2af_sw, b2_af, Wsaf_sw, Wvaf_sw,
      fs, fv);

  // ---- hf update ----
  post_kernel<<<N_FRAG / 64, 128, POST_LDS, stream>>>(
      N_FRAG, h_frag_s, h_frag_v, fs, fv, Wfs_sw, Wfv_sw, hf_s, hf_v);

  // ---- assemble output ----
  int total = N_ATOM * OUTC;
  assemble_kernel<<<(total + 255) / 256, 256, 0, stream>>>(ha_s, ha_v, hf_s, hf_v,
                                                           Ifin, frag_id, out);
  (void)in_sizes; (void)n_in; (void)out_size; (void)ws_size;
}
